// TitansMemoryModule_59863254172476
// MI455X (gfx1250) — compile-verified, matrix operand order checked by reading, not run on hardware
//
#include <hip/hip_runtime.h>

// ---------------------------------------------------------------------------
// Titans memory module on MI455X (gfx1250, wave32).
//   retrieved = k @ W^T            (WMMA bf16, fp32 acc)
//   loss      = mean(diff^2)
//   grad      = (2/numel) diff^T@k (WMMA bf16, fp32 acc, register-resident)
//   gates/momentum/norm-clip epilogue in a 1-block finalize kernel.
//
// Roofline: ~768 MB HBM @ 23.3 TB/s => ~33 us floor; 17.2 GFLOP via
// v_wmma_f32_16x16x32_bf16 finishes well inside that window.
// Wave-PAIR split over the j (output-column) dimension + tight operand
// scoping keeps the per-wave VGPR footprint in the 256 direct-address
// window (no s_set_vgpr_msb churn, better occupancy).
// ---------------------------------------------------------------------------

typedef __attribute__((ext_vector_type(16))) __bf16 v16bf;
typedef __attribute__((ext_vector_type(8)))  float  v8f;

#define WAVES_PER_BLOCK 8
#define ROWS_PER_TILE   32
#define MAIN_BLOCKS     1024

// ws layout (floats): [0] loss acc, [8..72) ksum acc, [128..128+4096) grad acc
#define WS_FLOATS 4224

__global__ void __launch_bounds__(256) zero_ws_kernel(float* ws) {
  int i = blockIdx.x * 256 + threadIdx.x;
  if (i < WS_FLOATS) ws[i] = 0.0f;
}

__global__ void __launch_bounds__(256) titans_main_kernel(
    const float* __restrict__ k, const float* __restrict__ v,
    const float* __restrict__ W, float* __restrict__ ret,
    float* __restrict__ ws, int Nrows)
{
  __shared__ float sGrad[4096];
  __shared__ float sKsum[64];
  __shared__ float sLoss[1];

  const int lane = threadIdx.x & 31;
  const int col  = lane & 15;            // N-index within 16-wide tile
  const int kofs = (lane < 16) ? 0 : 8;  // K packing offset (A/B operand layout)
  const int mofs = (lane >> 4) * 8;      // M offset for C/D layout

  const long waveId     = (long)blockIdx.x * WAVES_PER_BLOCK + (threadIdx.x >> 5);
  const long waveStride = (long)gridDim.x * WAVES_PER_BLOCK;
  const int  jhalf      = (int)(waveId & 1);        // this wave's j-half (0/1)
  const long pairId     = waveId >> 1;
  const long rowStep    = (waveStride >> 1) * ROWS_PER_TILE;

  // ---- Preload this wave's half of W as WMMA B operands (bf16) -------------
  // B[K=d][N=j] = W[j][d]; lane holds column j = 16*(2*jhalf+nt) + col,
  // K-halves {kofs..kofs+7, kofs+16..kofs+23} inside 32-wide kh window.
  v16bf Bop[2][2];
  #pragma unroll
  for (int nt = 0; nt < 2; ++nt) {
    #pragma unroll
    for (int kh = 0; kh < 2; ++kh) {
      const float* wrow = W + (size_t)((jhalf * 2 + nt) * 16 + col) * 64 + kh * 32;
      v16bf t;
      #pragma unroll
      for (int j = 0; j < 4; ++j) {      // packed pairs -> v_cvt_pk_bf16_f32
        t[2*j]     = (__bf16)wrow[kofs + 2*j];
        t[2*j + 1] = (__bf16)wrow[kofs + 2*j + 1];
        t[8 + 2*j]     = (__bf16)wrow[kofs + 16 + 2*j];
        t[8 + 2*j + 1] = (__bf16)wrow[kofs + 16 + 2*j + 1];
      }
      Bop[nt][kh] = t;
    }
  }

  const v8f zero8 = {0.f, 0.f, 0.f, 0.f, 0.f, 0.f, 0.f, 0.f};

  // Persistent grad accumulators: G[jt][dt] covers j in this wave's half.
  v8f G[2][4];
  #pragma unroll
  for (int jt = 0; jt < 2; ++jt)
    #pragma unroll
    for (int dt = 0; dt < 4; ++dt) G[jt][dt] = zero8;

  float lossAcc = 0.0f;
  float ksumAcc[4] = {0.f, 0.f, 0.f, 0.f};

  for (long r0 = pairId * ROWS_PER_TILE; r0 < (long)Nrows; r0 += rowStep) {
    // Prefetch next tile toward L2 (even wave: k, odd wave: v).
    if (r0 + rowStep < (long)Nrows) {
      if (jhalf == 0) {
        const float* pk = k + (size_t)(r0 + rowStep + col) * 64;
        __builtin_prefetch(pk, 0, 1);
        __builtin_prefetch(pk + 16 * 64, 0, 1);
      } else {
        __builtin_prefetch(v + (size_t)(r0 + rowStep + lane) * 64, 0, 1);
      }
    }

    // ---- GEMM1 + diff per 16-row subtile; diff^T operand built in place ----
    // Ad[jt]: A-layout for diff^T: lane holds row M=j, K = n-local
    // {0..7,16..23} (lanes<16) / {8..15,24..31} (lanes>=16) -- exactly this
    // lane's D-layout accumulator VGPRs of subtiles 0 and 1.
    v16bf Ad[2];
    #pragma unroll
    for (int st = 0; st < 2; ++st) {
      const float* krow = k + (size_t)(r0 + st * 16 + col) * 64;
      v16bf Aop[2];
      #pragma unroll
      for (int kh = 0; kh < 2; ++kh) {
        v16bf t;
        #pragma unroll
        for (int j = 0; j < 4; ++j) {
          t[2*j]     = (__bf16)krow[kh * 32 + kofs + 2*j];
          t[2*j + 1] = (__bf16)krow[kh * 32 + kofs + 2*j + 1];
          t[8 + 2*j]     = (__bf16)krow[kh * 32 + kofs + 16 + 2*j];
          t[8 + 2*j + 1] = (__bf16)krow[kh * 32 + kofs + 16 + 2*j + 1];
        }
        Aop[kh] = t;
      }
      #pragma unroll
      for (int nt = 0; nt < 2; ++nt) {
        v8f c = zero8;
        c = __builtin_amdgcn_wmma_f32_16x16x32_bf16(false, Aop[0], false, Bop[nt][0],
                                                    (short)0, c, false, false);
        c = __builtin_amdgcn_wmma_f32_16x16x32_bf16(false, Aop[1], false, Bop[nt][1],
                                                    (short)0, c, false, false);
        // diff = ret - v ; store retrieved ; loss ; fold into Ad immediately
        #pragma unroll
        for (int g = 0; g < 8; ++g) {
          size_t idx = (size_t)(r0 + st * 16 + mofs + g) * 64
                     + (jhalf * 2 + nt) * 16 + col;
          float rv = c[g];
          ret[idx] = rv;
          float d = rv - v[idx];
          c[g] = d;
          lossAcc = fmaf(d, d, lossAcc);
        }
        #pragma unroll
        for (int j = 0; j < 4; ++j) {
          Ad[nt][st*8 + 2*j]     = (__bf16)c[2*j];
          Ad[nt][st*8 + 2*j + 1] = (__bf16)c[2*j + 1];
        }
      }
    }

    // ---- GEMM2: G += diff^T @ k, one B operand (k tile, B-layout) at a time.
    // k re-read is WGP$/L2-resident (same rows were just streamed).
    #pragma unroll
    for (int dt = 0; dt < 4; ++dt) {
      v16bf Bk;
      float s = 0.0f;
      #pragma unroll
      for (int i = 0; i < 8; ++i) {
        float a = k[(size_t)(r0 + kofs + i) * 64 + dt * 16 + col];
        float b = k[(size_t)(r0 + kofs + 16 + i) * 64 + dt * 16 + col];
        Bk[i] = (__bf16)a; Bk[8 + i] = (__bf16)b;
        s += a + b;
      }
      if (jhalf == 0) ksumAcc[dt] += s;  // count each k element once globally
      #pragma unroll
      for (int jt = 0; jt < 2; ++jt)
        G[jt][dt] = __builtin_amdgcn_wmma_f32_16x16x32_bf16(false, Ad[jt], false, Bk,
                                                            (short)0, G[jt][dt], false, false);
    }
  }

  // ---- Block-level reduction through LDS, then global atomics --------------
  for (int i = threadIdx.x; i < 4096; i += 256) sGrad[i] = 0.0f;
  for (int i = threadIdx.x; i < 64;   i += 256) sKsum[i] = 0.0f;
  if (threadIdx.x == 0) sLoss[0] = 0.0f;
  __syncthreads();

  float l = lossAcc;
  #pragma unroll
  for (int off = 16; off > 0; off >>= 1) l += __shfl_xor(l, off, 32);
  if (lane == 0) atomicAdd(&sLoss[0], l);

  #pragma unroll
  for (int dt = 0; dt < 4; ++dt) {
    float s = ksumAcc[dt] + __shfl_xor(ksumAcc[dt], 16, 32);
    if (jhalf == 0 && lane < 16) atomicAdd(&sKsum[dt * 16 + col], s);
  }

  #pragma unroll
  for (int jt = 0; jt < 2; ++jt)
    #pragma unroll
    for (int dt = 0; dt < 4; ++dt)
      #pragma unroll
      for (int g = 0; g < 8; ++g) {
        int j = (jhalf * 2 + jt) * 16 + mofs + g;
        int d = dt * 16 + col;
        atomicAdd(&sGrad[j * 64 + d], G[jt][dt][g]);
      }
  __syncthreads();

  float* wsLoss = ws;
  float* wsKsum = ws + 8;
  float* wsGrad = ws + 128;
  if (threadIdx.x == 0) atomicAdd(wsLoss, sLoss[0]);
  for (int i = threadIdx.x; i < 64;   i += 256) atomicAdd(&wsKsum[i], sKsum[i]);
  for (int i = threadIdx.x; i < 4096; i += 256) atomicAdd(&wsGrad[i], sGrad[i]);
}

__global__ void __launch_bounds__(256) titans_finalize_kernel(
    const float* __restrict__ W, const float* __restrict__ gate_w,
    const float* __restrict__ gate_b, const float* __restrict__ S,
    const float* __restrict__ ws, float* __restrict__ out,
    long long retElems, int Nrows)
{
  __shared__ float sGates[3];
  __shared__ float sRed[256];
  __shared__ float sScale;

  const float* wsLoss = ws;
  const float* wsKsum = ws + 8;
  const float* wsGrad = ws + 128;

  const float numel = (float)Nrows * 64.0f;

  if (threadIdx.x < 3) {
    float acc = gate_b[threadIdx.x];
    for (int d = 0; d < 64; ++d)
      acc += (wsKsum[d] / (float)Nrows) * gate_w[threadIdx.x * 64 + d];
    sGates[threadIdx.x] = 1.0f / (1.0f + __expf(-acc));
  }
  __syncthreads();
  const float alpha = sGates[0], eta = sGates[1], theta = sGates[2];

  float* outLoss = out + retElems;
  float* outW    = outLoss + 1;
  float* outS    = outW + 4096;
  float* outG    = outS + 4096;

  if (threadIdx.x == 0) outLoss[0] = wsLoss[0] / numel;
  if (threadIdx.x < 3)  outG[threadIdx.x] = sGates[threadIdx.x];

  const float scaleG = 2.0f / numel;
  float local = 0.0f;
  for (int i = threadIdx.x; i < 4096; i += 256) {
    float g  = fminf(1.0f, fmaxf(-1.0f, wsGrad[i] * scaleG));
    float ns = eta * S[i] - 0.01f * theta * g;           // _TTT_LR_SCALE
    float nw = (1.0f - alpha) * W[i] + ns;
    outS[i] = ns;
    local += nw * nw;
  }
  sRed[threadIdx.x] = local;
  __syncthreads();
  for (int s = 128; s > 0; s >>= 1) {
    if (threadIdx.x < s) sRed[threadIdx.x] += sRed[threadIdx.x + s];
    __syncthreads();
  }
  if (threadIdx.x == 0) {
    float nrm = sqrtf(sRed[0]);
    sScale = (nrm > 10.0f) ? (10.0f / (nrm + 1e-8f)) : 1.0f;  // _MEMORY_MAX_NORM
  }
  __syncthreads();
  const float sc = sScale;
  for (int i = threadIdx.x; i < 4096; i += 256) {
    float g  = fminf(1.0f, fmaxf(-1.0f, wsGrad[i] * scaleG));
    float ns = eta * S[i] - 0.01f * theta * g;
    float nw = (1.0f - alpha) * W[i] + ns;
    outW[i] = nw * sc;
  }
}

extern "C" void kernel_launch(void* const* d_in, const int* in_sizes, int n_in,
                              void* d_out, int out_size, void* d_ws, size_t ws_size,
                              hipStream_t stream) {
  const float* k      = (const float*)d_in[0];
  const float* v      = (const float*)d_in[1];
  const float* W      = (const float*)d_in[2];
  const float* gate_w = (const float*)d_in[3];
  const float* gate_b = (const float*)d_in[4];
  const float* S      = (const float*)d_in[5];
  float* out = (float*)d_out;
  float* ws  = (float*)d_ws;

  const int Nrows = in_sizes[0] / 64;
  const long long retElems = (long long)Nrows * 64;

  zero_ws_kernel<<<(WS_FLOATS + 255) / 256, 256, 0, stream>>>(ws);
  titans_main_kernel<<<MAIN_BLOCKS, 256, 0, stream>>>(k, v, W, out, ws, Nrows);
  titans_finalize_kernel<<<1, 256, 0, stream>>>(W, gate_w, gate_b, S, ws, out, retElems, Nrows);
}
